// GCN_21139829031083
// MI455X (gfx1250) — compile-verified, matrix operand order checked by reading, not run on hardware
//
#include <hip/hip_runtime.h>
#include <hip/hip_bf16.h>

typedef __attribute__((ext_vector_type(16))) _Float16 v16h;
typedef __attribute__((ext_vector_type(8)))  _Float16 v8h;
typedef __attribute__((ext_vector_type(8)))  float    v8f;

// ---------------------------------------------------------------- utilities

__global__ void k_fill(float* __restrict__ p, float v, int n) {
  int i = blockIdx.x * blockDim.x + threadIdx.x;
  if (i < n) p[i] = v;
}

// deg[dst] += 1 for each real edge (self-loop contribution already in init=1.0)
__global__ void k_deg(const int* __restrict__ dst, float* __restrict__ deg, int E) {
  int e = blockIdx.x * blockDim.x + threadIdx.x;
  if (e < E) atomicAdd(&deg[dst[e]], 1.0f);
}

__global__ void k_rsqrt(float* __restrict__ p, int n) {
  int i = blockIdx.x * blockDim.x + threadIdx.x;
  if (i < n) p[i] = rsqrtf(p[i]);   // deg >= 1 always (self loops)
}

// out[i*Fdim + f] = b[f]  (bias broadcast as scatter-add init)
__global__ void k_bias_init(float* __restrict__ out, const float* __restrict__ b,
                            int total, int Fdim) {
  int i = blockIdx.x * blockDim.x + threadIdx.x;
  if (i < total) out[i] = b[i & (Fdim - 1)];   // Fdim is 64 or 32 (pow2)
}

// ------------------------------------------------------- WMMA fragment loads
// 16-bit A-matrix 16x32 layout (ISA 7.12.2): lane(l): row = l&15, half = l>=16.
// element i of v16h holds K = (i&7) + 2*(i&8) + 8*half  (runs [0..7] and [16..23]).

template <bool RELU>
__device__ __forceinline__ v16h load_a_frag(const float* __restrict__ rowp,
                                            int k0, int half) {
  const float* p = rowp + k0 + half * 8;
  float4 u0 = *(const float4*)(p);
  float4 u1 = *(const float4*)(p + 4);
  float4 u2 = *(const float4*)(p + 16);
  float4 u3 = *(const float4*)(p + 20);
  if (RELU) {
    u0.x = fmaxf(u0.x, 0.f); u0.y = fmaxf(u0.y, 0.f); u0.z = fmaxf(u0.z, 0.f); u0.w = fmaxf(u0.w, 0.f);
    u1.x = fmaxf(u1.x, 0.f); u1.y = fmaxf(u1.y, 0.f); u1.z = fmaxf(u1.z, 0.f); u1.w = fmaxf(u1.w, 0.f);
    u2.x = fmaxf(u2.x, 0.f); u2.y = fmaxf(u2.y, 0.f); u2.z = fmaxf(u2.z, 0.f); u2.w = fmaxf(u2.w, 0.f);
    u3.x = fmaxf(u3.x, 0.f); u3.y = fmaxf(u3.y, 0.f); u3.z = fmaxf(u3.z, 0.f); u3.w = fmaxf(u3.w, 0.f);
  }
  v16h a;
  a[0]  = (_Float16)u0.x; a[1]  = (_Float16)u0.y; a[2]  = (_Float16)u0.z; a[3]  = (_Float16)u0.w;
  a[4]  = (_Float16)u1.x; a[5]  = (_Float16)u1.y; a[6]  = (_Float16)u1.z; a[7]  = (_Float16)u1.w;
  a[8]  = (_Float16)u2.x; a[9]  = (_Float16)u2.y; a[10] = (_Float16)u2.z; a[11] = (_Float16)u2.w;
  a[12] = (_Float16)u3.x; a[13] = (_Float16)u3.y; a[14] = (_Float16)u3.z; a[15] = (_Float16)u3.w;
  return a;
}

// B fragment from LDS holding W^T as f16 ([n][k], row length = K dim).
__device__ __forceinline__ v16h load_b_frag(const _Float16* __restrict__ colp,
                                            int k0, int half) {
  const _Float16* p = colp + k0 + half * 8;
  v8h b0 = *(const v8h*)(p);        // K run +0..+7
  v8h b1 = *(const v8h*)(p + 16);   // K run +16..+23
  return __builtin_shufflevector(b0, b1, 0,1,2,3,4,5,6,7,8,9,10,11,12,13,14,15);
}

// ------------------------------------------------------------------- GEMM 1
// xw[N,64] = x[N,256] @ W1[256,64].
// Block = 256 thr (8 waves); each wave owns one 16-row M-tile and computes
// ALL 4 N-tiles, so each A load + f32->f16 convert feeds 4 WMMAs.
__global__ __launch_bounds__(256)
void k_gemm1(const float* __restrict__ x, const float* __restrict__ W,
             float* __restrict__ xw, int Nn) {
  constexpr int F = 256, H = 64;
  __shared__ _Float16 sW[H * F];              // W^T as f16: sW[n*F + k], 32 KB
  int tid = threadIdx.x;
  for (int i = tid; i < F * H; i += 256) {
    int k = i / H, n = i % H;
    sW[n * F + k] = (_Float16)W[i];
  }
  __syncthreads();

  int wave = tid >> 5, lane = tid & 31;
  int half = lane >> 4, lx = lane & 15;
  int mt = blockIdx.x * 8 + wave;             // M tile (16 rows) per wave

  int arow = mt * 16 + lx;
  if (arow >= Nn) arow = Nn - 1;              // clamp, keep EXEC uniform
  const float* ap = x + (size_t)arow * F;

  v8f acc[4] = {v8f{}, v8f{}, v8f{}, v8f{}};
#pragma unroll
  for (int k0 = 0; k0 < F; k0 += 32) {
    v16h a = load_a_frag<false>(ap, k0, half);
#pragma unroll
    for (int nt = 0; nt < 4; ++nt) {
      v16h b = load_b_frag(sW + (size_t)(nt * 16 + lx) * F, k0, half);
      acc[nt] = __builtin_amdgcn_wmma_f32_16x16x32_f16(false, a, false, b,
                                                       (short)0, acc[nt], false, false);
    }
  }

  int mbase = mt * 16 + half * 8;
#pragma unroll
  for (int nt = 0; nt < 4; ++nt) {
    int col = nt * 16 + lx;
#pragma unroll
    for (int r = 0; r < 8; ++r) {
      int row = mbase + r;
      if (row < Nn) xw[(size_t)row * H + col] = acc[nt][r];
    }
  }
}

// ------------------------------------------------------------------- GEMM 2
// hw[N,32] = relu(h)[N,64] @ W2[64,32].  One M-tile per wave, 2 N-tiles.
__global__ __launch_bounds__(256)
void k_gemm2(const float* __restrict__ h, const float* __restrict__ W,
             float* __restrict__ hw, int Nn) {
  constexpr int K = 64, C = 32;
  __shared__ _Float16 sW[C * K];              // W^T as f16: sW[n*K + k], 4 KB
  int tid = threadIdx.x;
  for (int i = tid; i < K * C; i += 256) {
    int k = i / C, n = i % C;
    sW[n * K + k] = (_Float16)W[i];
  }
  __syncthreads();

  int wave = tid >> 5, lane = tid & 31;
  int half = lane >> 4, lx = lane & 15;
  int mt = blockIdx.x * 8 + wave;

  int arow = mt * 16 + lx;
  if (arow >= Nn) arow = Nn - 1;
  const float* ap = h + (size_t)arow * K;

  v8f acc[2] = {v8f{}, v8f{}};
#pragma unroll
  for (int k0 = 0; k0 < K; k0 += 32) {
    v16h a = load_a_frag<true>(ap, k0, half);   // fused ReLU on layer-1 output
#pragma unroll
    for (int nt = 0; nt < 2; ++nt) {
      v16h b = load_b_frag(sW + (size_t)(nt * 16 + lx) * K, k0, half);
      acc[nt] = __builtin_amdgcn_wmma_f32_16x16x32_f16(false, a, false, b,
                                                       (short)0, acc[nt], false, false);
    }
  }

  int mbase = mt * 16 + half * 8;
#pragma unroll
  for (int nt = 0; nt < 2; ++nt) {
    int col = nt * 16 + lx;
#pragma unroll
    for (int r = 0; r < 8; ++r) {
      int row = mbase + r;
      if (row < Nn) hw[(size_t)row * C + col] = acc[nt][r];
    }
  }
}

// ------------------------------------------------------------------ scatter
// out[dst] += msg[src] * dis[src]*dis[dst], over E edges + N self loops.
// Fdim/4 threads per edge; float4 gather (L2-resident), f32 atomics into L2.
__global__ void k_scatter(const int* __restrict__ src, const int* __restrict__ dst,
                          const float* __restrict__ dis, const float* __restrict__ msg,
                          float* __restrict__ out, int E, int Nn, int Fdim) {
  long long t = (long long)blockIdx.x * blockDim.x + threadIdx.x;
  int tpe = Fdim >> 2;                       // threads per edge
  long long e = t / tpe;
  int f = (int)(t - e * tpe) * 4;
  if (e >= (long long)E + Nn) return;
  int s, d;
  if (e < E) { s = src[e]; d = dst[e]; }
  else       { s = d = (int)(e - E); }       // self loop
  float nrm = dis[s] * dis[d];
  float4 m = *(const float4*)(msg + (size_t)s * Fdim + f);
  float* o = out + (size_t)d * Fdim + f;
  atomicAdd(o + 0, m.x * nrm);
  atomicAdd(o + 1, m.y * nrm);
  atomicAdd(o + 2, m.z * nrm);
  atomicAdd(o + 3, m.w * nrm);
}

// ------------------------------------------------------------------- driver

extern "C" void kernel_launch(void* const* d_in, const int* in_sizes, int n_in,
                              void* d_out, int out_size, void* d_ws, size_t ws_size,
                              hipStream_t stream) {
  const float* x  = (const float*)d_in[0];
  const int*   ei = (const int*)  d_in[1];
  const float* W1 = (const float*)d_in[2];
  const float* b1 = (const float*)d_in[3];
  const float* W2 = (const float*)d_in[4];
  const float* b2 = (const float*)d_in[5];

  const int H  = in_sizes[3];            // 64
  const int C  = in_sizes[5];            // 32
  const int F  = in_sizes[2] / H;        // 256
  const int Nn = in_sizes[0] / F;        // 100000
  const int E  = in_sizes[1] / 2;        // 1.6M
  const int* srcv = ei;
  const int* dstv = ei + E;

  // workspace layout (~64.5 MB): dis | xw | h1 | hw2
  char* ws = (char*)d_ws;
  size_t off = 0;
  float* dis = (float*)(ws + off); off += (((size_t)Nn * 4) + 255) & ~(size_t)255;
  float* xw  = (float*)(ws + off); off += (((size_t)Nn * H * 4) + 255) & ~(size_t)255;
  float* h1  = (float*)(ws + off); off += (((size_t)Nn * H * 4) + 255) & ~(size_t)255;
  float* hw2 = (float*)(ws + off);
  float* outp = (float*)d_out;

  // 1) degree -> deg^{-1/2}
  k_fill <<<(Nn + 255) / 256, 256, 0, stream>>>(dis, 1.0f, Nn);
  k_deg  <<<(E  + 255) / 256, 256, 0, stream>>>(dstv, dis, E);
  k_rsqrt<<<(Nn + 255) / 256, 256, 0, stream>>>(dis, Nn);

  // 2) layer 1: xw = x@W1 ; h1 = b1 + scatter(xw * norm)
  k_bias_init<<<((Nn * H) + 255) / 256, 256, 0, stream>>>(h1, b1, Nn * H, H);
  k_gemm1<<<(Nn + 127) / 128, 256, 0, stream>>>(x, W1, xw, Nn);
  {
    long long total = (long long)(E + Nn) * (H / 4);
    k_scatter<<<(int)((total + 255) / 256), 256, 0, stream>>>(srcv, dstv, dis, xw, h1, E, Nn, H);
  }

  // 3) layer 2: hw2 = relu(h1)@W2 ; out = b2 + scatter(hw2 * norm)
  k_gemm2<<<(Nn + 127) / 128, 256, 0, stream>>>(h1, W2, hw2, Nn);
  k_bias_init<<<((Nn * C) + 255) / 256, 256, 0, stream>>>(outp, b2, Nn * C, C);
  {
    long long total = (long long)(E + Nn) * (C / 4);
    k_scatter<<<(int)((total + 255) / 256), 256, 0, stream>>>(srcv, dstv, dis, hw2, outp, E, Nn, C);
  }
}
